// AttentionBlock_81389630259580
// MI455X (gfx1250) — compile-verified, matrix operand order checked by reading, not run on hardware
//
#include <hip/hip_runtime.h>
#include <hip/hip_bf16.h>

// ---------------------------------------------------------------------------
// CDNA5 (gfx1250) fused attention block, bf16 WMMA pipeline + TDM staging.
// wave32; WMMA 16x16x32 bf16 -> f32 accumulate; tensor_load_to_lds for tiles.
// ---------------------------------------------------------------------------

typedef __attribute__((ext_vector_type(16))) __bf16 v16bf;
typedef __attribute__((ext_vector_type(8)))  __bf16 v8bf;
typedef __attribute__((ext_vector_type(8)))  float  v8f;
typedef __attribute__((ext_vector_type(4)))  unsigned int v4ui;
typedef __attribute__((ext_vector_type(8)))  int    v8si;
typedef __attribute__((ext_vector_type(4)))  int    v4si;

#define DMODEL 768
#define SEQ    2048
#define NBATCH 8
#define NTOK   (NBATCH * SEQ)

__device__ __forceinline__ v8f wmma_bf16(v16bf a, v16bf b, v8f c) {
  return __builtin_amdgcn_wmma_f32_16x16x32_bf16(false, a, false, b, (short)0, c,
                                                 false, false);
}

// Load a 16x32 bf16 "A-style" fragment from row-major memory (stride in elems).
// Lanes 0-15: row M=lane, K=0..7 (v0..3), K=16..23 (v4..7);
// lanes 16-31: row M=lane-16, K=8..15, K=24..31 -> two contiguous 16B chunks.
__device__ __forceinline__ v16bf load_frag(const __bf16* base, int stride, int lane) {
  const int half = (lane >> 4) & 1;
  const int row  = lane & 15;
  const __bf16* p = base + (size_t)row * stride + half * 8;
  v8bf lo = *(const v8bf*)(p);
  v8bf hi = *(const v8bf*)(p + 16);
  return __builtin_shufflevector(lo, hi, 0, 1, 2, 3, 4, 5, 6, 7,
                                 8, 9, 10, 11, 12, 13, 14, 15);
}

// ---------------------------------------------------------------------------
// Tensor Data Mover: 2-D tile load (global -> LDS), bf16 elements.
// D# per CDNA5 ISA:
//   group0: [1:0]=count=1, [63:32]=lds_addr, [120:64]=global_addr, [127:126]=2
//   group1: [17:16]=data_size(1 -> 2B), [20]=pad_en, [24:22]=pad_interval,
//           [31:25]=pad_amount, [79:48]=tensor_dim0, [111:80]=tensor_dim1,
//           [127:112]=tile_dim0, [143:128]=tile_dim1, [207:160]=dim0 stride.
// pad_interval=4 (32 DWORDs = one 64-elem row), pad_amount=3 (4 DWORDs = 16B)
// reproduces the padded 72-element LDS row stride used by load_frag.
// ---------------------------------------------------------------------------
__device__ __forceinline__ void tdm_load_2d(unsigned int lds_off, const void* gptr,
                                            unsigned int tile_d0, unsigned int tile_d1,
                                            unsigned int tensor_d0, unsigned int tensor_d1,
                                            unsigned int stride_e, int pad_interval,
                                            int pad_amount, bool pad_en) {
  const unsigned long long ga = (unsigned long long)(uintptr_t)gptr;
  v4ui g0;
  g0.x = 1u;                                   // count=1, user descriptor
  g0.y = lds_off;                              // LDS byte address
  g0.z = (unsigned int)ga;                     // global addr [31:0]
  g0.w = (unsigned int)((ga >> 32) & 0x01FFFFFFu) | (2u << 30);  // [56:32]|type=2

  union { unsigned long long q[4]; v8si v; } g1;
  unsigned int w0 = (1u << 16);                // data_size = 2 bytes
  if (pad_en)
    w0 |= (1u << 20) | ((unsigned)pad_interval << 22) | ((unsigned)pad_amount << 25);
  g1.q[0] = (unsigned long long)w0 |
            ((unsigned long long)(tensor_d0 & 0xFFFFu) << 48);
  g1.q[1] = ((unsigned long long)tensor_d0 >> 16) |
            ((unsigned long long)tensor_d1 << 16) |
            ((unsigned long long)(tile_d0 & 0xFFFFu) << 48);
  g1.q[2] = (unsigned long long)(tile_d1 & 0xFFFFu) |
            ((unsigned long long)stride_e << 32);
  g1.q[3] = 0ull;

  v4si z4 = {0, 0, 0, 0};
#if __clang_major__ >= 23
  v8si z8 = {0, 0, 0, 0, 0, 0, 0, 0};
  __builtin_amdgcn_tensor_load_to_lds(g0, g1.v, z4, z4, z8, 0);
#else
  __builtin_amdgcn_tensor_load_to_lds(g0, g1.v, z4, z4, 0);
#endif
}

// ---------------------------------------------------------------------------
// f32 -> bf16 conversion (plain, and transposing variant for weights)
// ---------------------------------------------------------------------------
__global__ void cvt_bf16_kernel(const float* __restrict__ in,
                                __bf16* __restrict__ out, int n) {
  int i = blockIdx.x * 256 + threadIdx.x;
  if (i < n) out[i] = (__bf16)in[i];
}

// out[n*K + k] = (bf16) in[k*N + n]   (W [K][N] -> Wt [N][K])
__global__ void cvt_bf16_t_kernel(const float* __restrict__ in,
                                  __bf16* __restrict__ out, int K, int N) {
  int i = blockIdx.x * 256 + threadIdx.x;
  if (i < K * N) {
    int n = i / K, k = i % K;
    out[i] = (__bf16)in[(size_t)k * N + n];
  }
}

// ---------------------------------------------------------------------------
// GEMM: C[M,N] = A[M,K](bf16) * Bt[N,K](bf16, pre-transposed) + bias
// MODE 0: f32 out (+bias)
// MODE 1: bf16 out, relu(acc+bias)               (Q, K projections)
// MODE 2: bf16 out, relu, transposed per batch:  out[(b*DMODEL+n)*SEQ + s]
// Tiles: BM=128, BN=128, KT=64; 8 waves (2x4); TDM double-buffered staging.
// ---------------------------------------------------------------------------
template <int MODE>
__global__ __launch_bounds__(256) void gemm_bf16_kernel(
    const __bf16* __restrict__ A, const __bf16* __restrict__ Bt,
    const float* __restrict__ bias, float* __restrict__ Cf,
    __bf16* __restrict__ Cb, int M, int N, int K) {
  const int SA = 72;  // 64 data + 8 pad elems (144B rows, via TDM padding)
  __shared__ __bf16 As[2][128 * 72];
  __shared__ __bf16 Bs[2][128 * 72];

  const int t    = threadIdx.x;
  const int lane = t & 31;
  const int wave = t >> 5;
  const int wr   = wave >> 2;  // 0..1  (rows of 64)
  const int wc   = wave & 3;   // 0..3  (cols of 32)
  const int m0   = blockIdx.y * 128;
  const int n0   = blockIdx.x * 128;

  const unsigned int asOff[2] = {(unsigned int)(uintptr_t)&As[0][0],
                                 (unsigned int)(uintptr_t)&As[1][0]};
  const unsigned int bsOff[2] = {(unsigned int)(uintptr_t)&Bs[0][0],
                                 (unsigned int)(uintptr_t)&Bs[1][0]};

  v8f acc[4][2];
#pragma unroll
  for (int i = 0; i < 4; i++)
#pragma unroll
    for (int j = 0; j < 2; j++) acc[i][j] = {};

  const int kIter = K / 64;
  if (wave == 0) {
    tdm_load_2d(asOff[0], A + (size_t)m0 * K, 64, 128, K, M, K, 4, 3, true);
    tdm_load_2d(bsOff[0], Bt + (size_t)n0 * K, 64, 128, K, N, K, 4, 3, true);
  }

  for (int i = 0; i < kIter; i++) {
    if (wave == 0) {
      if (i + 1 < kIter) {
        const int k0 = (i + 1) * 64;
        tdm_load_2d(asOff[(i + 1) & 1], A + (size_t)m0 * K + k0, 64, 128, K, M,
                    K, 4, 3, true);
        tdm_load_2d(bsOff[(i + 1) & 1], Bt + (size_t)n0 * K + k0, 64, 128, K, N,
                    K, 4, 3, true);
        __builtin_amdgcn_s_wait_tensorcnt(2);  // current pair complete
      } else {
        __builtin_amdgcn_s_wait_tensorcnt(0);
      }
    }
    __syncthreads();
    const __bf16* Ab = &As[i & 1][0];
    const __bf16* Bb = &Bs[i & 1][0];
#pragma unroll
    for (int kk = 0; kk < 64; kk += 32) {
      v16bf af[4], bfr[2];
#pragma unroll
      for (int mt = 0; mt < 4; mt++)
        af[mt] = load_frag(Ab + (wr * 64 + mt * 16) * SA + kk, SA, lane);
#pragma unroll
      for (int nt = 0; nt < 2; nt++)
        bfr[nt] = load_frag(Bb + (wc * 32 + nt * 16) * SA + kk, SA, lane);
#pragma unroll
      for (int mt = 0; mt < 4; mt++)
#pragma unroll
        for (int nt = 0; nt < 2; nt++)
          acc[mt][nt] = wmma_bf16(af[mt], bfr[nt], acc[mt][nt]);
    }
    __syncthreads();
  }

  const int half = lane >> 4, col = lane & 15;
#pragma unroll
  for (int mt = 0; mt < 4; mt++) {
#pragma unroll
    for (int nt = 0; nt < 2; nt++) {
      const int n     = n0 + wc * 32 + nt * 16 + col;
      const float bv  = bias ? bias[n] : 0.0f;
      const int mbase = m0 + wr * 64 + mt * 16;
      if (MODE == 2) {
        const int b = mbase / SEQ;
        const int s = (mbase % SEQ) + half * 8;
        v8bf ov;
#pragma unroll
        for (int v = 0; v < 8; v++) {
          float x = acc[mt][nt][v] + bv;
          x       = x > 0.f ? x : 0.f;
          ov[v]   = (__bf16)x;
        }
        *(v8bf*)(Cb + ((size_t)b * DMODEL + n) * SEQ + s) = ov;
      } else {
#pragma unroll
        for (int v = 0; v < 8; v++) {
          const int m = mbase + v + half * 8;
          float x     = acc[mt][nt][v] + bv;
          if (MODE == 1) {
            x = x > 0.f ? x : 0.f;
            Cb[(size_t)m * N + n] = (__bf16)x;
          } else {
            Cf[(size_t)m * N + n] = x;
          }
        }
      }
    }
  }
}

// ---------------------------------------------------------------------------
// Attention: per WG = 16 query rows of one batch.
// ---------------------------------------------------------------------------
__global__ __launch_bounds__(256) void attn_kernel(
    const __bf16* __restrict__ Qb, const __bf16* __restrict__ Kb,
    const __bf16* __restrict__ Vt, float* __restrict__ O) {
  extern __shared__ char smem[];
  __bf16* Qs = (__bf16*)smem;                                   // 16*768
  float*  Sb = (float*)(smem + 16 * 768 * 2);                   // 16*2048 f32
  __bf16* Pb = (__bf16*)(smem + 16 * 768 * 2 + 16 * 2048 * 4);  // 16*2048
  float*  red =
      (float*)(smem + 16 * 768 * 2 + 16 * 2048 * 4 + 16 * 2048 * 2);
  float* rowmax = red;       // 16
  float* rowsum = red + 16;  // 16
  float* part   = red + 32;  // 256

  const int t = threadIdx.x, lane = t & 31, w = t >> 5;
  const int qt = blockIdx.x, b = blockIdx.y;
  const size_t qrow0 = (size_t)b * SEQ + (size_t)qt * 16;

  // Q tile [16,768] via Tensor Data Mover (wave0 issues, barrier publishes).
  if (w == 0) {
    tdm_load_2d((unsigned int)(uintptr_t)Qs, Qb + qrow0 * DMODEL, DMODEL, 16,
                DMODEL, NTOK, DMODEL, 0, 0, false);
    __builtin_amdgcn_s_wait_tensorcnt(0);
  }
  __syncthreads();

  // ---- phase 1: S = Q K^T -------------------------------------------------
  {
    v8f sacc[16];
#pragma unroll
    for (int i = 0; i < 16; i++) sacc[i] = {};
    for (int kd = 0; kd < DMODEL; kd += 32) {
      v16bf af = load_frag(Qs + kd, DMODEL, lane);
#pragma unroll
      for (int ct = 0; ct < 16; ct++) {
        const __bf16* kp =
            Kb + ((size_t)b * SEQ + w * 256 + ct * 16) * DMODEL + kd;
        v16bf bfr = load_frag(kp, DMODEL, lane);
        sacc[ct]  = wmma_bf16(af, bfr, sacc[ct]);
      }
    }
    const int half = lane >> 4, col = lane & 15;
#pragma unroll
    for (int ct = 0; ct < 16; ct++)
#pragma unroll
      for (int v = 0; v < 8; v++)
        Sb[(v + half * 8) * SEQ + w * 256 + ct * 16 + col] = sacc[ct][v];
  }
  __syncthreads();

  // ---- phase 2: softmax ---------------------------------------------------
  {
    const int row = t >> 4, sub = t & 15;
    float m = -3.0e38f;
    for (int j = 0; j < 128; j++) m = fmaxf(m, Sb[row * SEQ + sub + j * 16]);
    part[t] = m;
    __syncthreads();
    if (sub == 0) {
      float mm = part[row * 16];
      for (int j = 1; j < 16; j++) mm = fmaxf(mm, part[row * 16 + j]);
      rowmax[row] = mm;
    }
    __syncthreads();
    const float mm = rowmax[row];
    float s = 0.f;
    for (int j = 0; j < 128; j++) {
      int idx = row * SEQ + sub + j * 16;
      float e = __expf(Sb[idx] - mm);
      Pb[idx] = (__bf16)e;
      s += e;
    }
    part[t] = s;
    __syncthreads();
    if (sub == 0) {
      float ss = 0.f;
      for (int j = 0; j < 16; j++) ss += part[row * 16 + j];
      rowsum[row] = ss;
    }
  }
  __syncthreads();

  // ---- phase 3: O = P V ---------------------------------------------------
  {
    v8f oacc[6];
#pragma unroll
    for (int i = 0; i < 6; i++) oacc[i] = {};
    for (int kt = 0; kt < SEQ; kt += 32) {
      v16bf af = load_frag(Pb + kt, SEQ, lane);
#pragma unroll
      for (int nt = 0; nt < 6; nt++) {
        const __bf16* vp =
            Vt + ((size_t)b * DMODEL + w * 96 + nt * 16) * SEQ + kt;
        v16bf bfr = load_frag(vp, SEQ, lane);
        oacc[nt]  = wmma_bf16(af, bfr, oacc[nt]);
      }
    }
    const int half = lane >> 4, col = lane & 15;
#pragma unroll
    for (int nt = 0; nt < 6; nt++) {
      const int n = w * 96 + nt * 16 + col;
#pragma unroll
      for (int v = 0; v < 8; v++) {
        const int r = v + half * 8;
        O[(qrow0 + r) * DMODEL + n] = oacc[nt][v] * (1.0f / rowsum[r]);
      }
    }
  }
}

// ---------------------------------------------------------------------------
// Residual + LayerNorm: out = LN(X + R) * g + beta; optional bf16 copy.
// ---------------------------------------------------------------------------
__global__ __launch_bounds__(256) void ln_kernel(
    const float* __restrict__ X, const float* __restrict__ R,
    const float* __restrict__ g, const float* __restrict__ beta,
    float* __restrict__ outF, __bf16* __restrict__ outB) {
  __shared__ float sbuf[256];
  const int row = blockIdx.x, t = threadIdx.x;
  float x[3];
  float s = 0.f;
#pragma unroll
  for (int j = 0; j < 3; j++) {
    int i = t + j * 256;
    x[j]  = X[(size_t)row * 768 + i] + R[(size_t)row * 768 + i];
    s += x[j];
  }
  sbuf[t] = s;
  __syncthreads();
  for (int st = 128; st > 0; st >>= 1) {
    if (t < st) sbuf[t] += sbuf[t + st];
    __syncthreads();
  }
  const float mu = sbuf[0] * (1.0f / 768.0f);
  __syncthreads();
  float v = 0.f;
#pragma unroll
  for (int j = 0; j < 3; j++) {
    float d = x[j] - mu;
    v += d * d;
  }
  sbuf[t] = v;
  __syncthreads();
  for (int st = 128; st > 0; st >>= 1) {
    if (t < st) sbuf[t] += sbuf[t + st];
    __syncthreads();
  }
  const float var = sbuf[0] * (1.0f / 768.0f);
  const float rs  = rsqrtf(var + 1e-5f);
#pragma unroll
  for (int j = 0; j < 3; j++) {
    int i   = t + j * 256;
    float y = (x[j] - mu) * rs * g[i] + beta[i];
    outF[(size_t)row * 768 + i] = y;
    if (outB) outB[(size_t)row * 768 + i] = (__bf16)y;
  }
}

// ---------------------------------------------------------------------------
extern "C" void kernel_launch(void* const* d_in, const int* in_sizes, int n_in,
                              void* d_out, int out_size, void* d_ws,
                              size_t ws_size, hipStream_t stream) {
  (void)in_sizes; (void)n_in; (void)out_size; (void)ws_size;
  const float* X  = (const float*)d_in[0];
  const float* Wq = (const float*)d_in[1];
  const float* bq = (const float*)d_in[2];
  const float* Wk = (const float*)d_in[3];
  const float* bk = (const float*)d_in[4];
  const float* Wv = (const float*)d_in[5];
  const float* bv = (const float*)d_in[6];
  const float* Wd = (const float*)d_in[7];
  const float* bd = (const float*)d_in[8];
  const float* g1 = (const float*)d_in[9];
  const float* b1 = (const float*)d_in[10];
  const float* g2 = (const float*)d_in[11];
  const float* b2 = (const float*)d_in[12];

  char*  ws  = (char*)d_ws;
  size_t off = 0;
  auto alloc = [&](size_t bytes) {
    size_t o = off;
    off += (bytes + 255) & ~(size_t)255;
    return o;
  };
  __bf16* Xb  = (__bf16*)(ws + alloc((size_t)NTOK * DMODEL * 2));
  __bf16* Wqt = (__bf16*)(ws + alloc((size_t)DMODEL * DMODEL * 2));
  __bf16* Wkt = (__bf16*)(ws + alloc((size_t)DMODEL * DMODEL * 2));
  __bf16* Wvt = (__bf16*)(ws + alloc((size_t)DMODEL * DMODEL * 2));
  __bf16* Wdt = (__bf16*)(ws + alloc((size_t)DMODEL * DMODEL * 2));
  __bf16* Qb  = (__bf16*)(ws + alloc((size_t)NTOK * DMODEL * 2));
  __bf16* Kb  = (__bf16*)(ws + alloc((size_t)NTOK * DMODEL * 2));
  __bf16* Vtb = (__bf16*)(ws + alloc((size_t)NTOK * DMODEL * 2));  // [b][d][s]
  float*  attnF = (float*)(ws + alloc((size_t)NTOK * DMODEL * 4));
  float*  X1    = (float*)(ws + alloc((size_t)NTOK * DMODEL * 4));
  __bf16* X1b = Qb;     // Q is dead after attention
  float*  Yf  = attnF;  // attn is dead after LN1

  const int nX = NTOK * DMODEL;
  const int nW = DMODEL * DMODEL;

  cvt_bf16_kernel<<<(nX + 255) / 256, 256, 0, stream>>>(X, Xb, nX);
  cvt_bf16_t_kernel<<<(nW + 255) / 256, 256, 0, stream>>>(Wq, Wqt, DMODEL, DMODEL);
  cvt_bf16_t_kernel<<<(nW + 255) / 256, 256, 0, stream>>>(Wk, Wkt, DMODEL, DMODEL);
  cvt_bf16_t_kernel<<<(nW + 255) / 256, 256, 0, stream>>>(Wv, Wvt, DMODEL, DMODEL);
  cvt_bf16_t_kernel<<<(nW + 255) / 256, 256, 0, stream>>>(Wd, Wdt, DMODEL, DMODEL);

  dim3 ggrid(DMODEL / 128, NTOK / 128);  // (6, 128)
  gemm_bf16_kernel<1><<<ggrid, 256, 0, stream>>>(Xb, Wqt, bq, nullptr, Qb,
                                                 NTOK, DMODEL, DMODEL);
  gemm_bf16_kernel<1><<<ggrid, 256, 0, stream>>>(Xb, Wkt, bk, nullptr, Kb,
                                                 NTOK, DMODEL, DMODEL);
  gemm_bf16_kernel<2><<<ggrid, 256, 0, stream>>>(Xb, Wvt, bv, nullptr, Vtb,
                                                 NTOK, DMODEL, DMODEL);

  const size_t ATTN_SMEM = (size_t)16 * 768 * 2 + (size_t)16 * 2048 * 4 +
                           (size_t)16 * 2048 * 2 + (size_t)(32 + 256) * 4;
  attn_kernel<<<dim3(SEQ / 16, NBATCH), 256, ATTN_SMEM, stream>>>(Qb, Kb, Vtb,
                                                                  attnF);

  ln_kernel<<<NTOK, 256, 0, stream>>>(X, attnF, g1, b1, X1, X1b);

  gemm_bf16_kernel<0><<<ggrid, 256, 0, stream>>>(X1b, Wdt, bd, Yf, nullptr,
                                                 NTOK, DMODEL, DMODEL);

  ln_kernel<<<NTOK, 256, 0, stream>>>(X1, Yf, g2, b2, (float*)d_out, nullptr);
}
